// JointModel_17652315587026
// MI455X (gfx1250) — matrix operand order repaired and checked by reading, not compiled, not run on hardware
//
#include <hip/hip_runtime.h>
#include <cmath>

// ---------------------------------------------------------------------------
// MI455X / gfx1250 implementation. wave32, WMMA bf16 16x16x32 for all GEMMs,
// double-buffered LDS staging with async global->LDS DMA (ASYNCcnt path),
// two barriers per K-step, global_prefetch for the weight stream.
// ---------------------------------------------------------------------------

typedef __bf16 bf16_t;
typedef __attribute__((ext_vector_type(16))) __bf16 v16bf;
typedef __attribute__((ext_vector_type(8)))  float  v8f;

#define B_   2
#define T_   545
#define TP_  576      // T padded to multiple of 64 for score/prob tiles
#define NH_  8
#define HD_  256
#define L_   2

// gfx1250 async global->LDS staging (GLOBAL_LOAD_ASYNC_TO_LDS_B128).
#define USE_ASYNC_LDS 1

static __device__ inline void wait_asynccnt0() {
#if defined(__has_builtin) && __has_builtin(__builtin_amdgcn_s_wait_asynccnt)
  __builtin_amdgcn_s_wait_asynccnt(0);
#else
  asm volatile("s_wait_asynccnt 0" ::: "memory");
#endif
}

static __device__ inline void async_copy16_to_lds(const bf16_t* gp, bf16_t* lp) {
  // VDST = LDS byte address (low 32 bits of generic shared pointer),
  // VADDR = 64-bit global address, no SADDR.
  unsigned lds_off = (unsigned)(size_t)lp;
  unsigned long long gaddr = (unsigned long long)(size_t)gp;
  asm volatile("global_load_async_to_lds_b128 %0, %1, off"
               :: "v"(lds_off), "v"(gaddr)
               : "memory");
}

static __device__ inline v8f zero_v8f() {
  v8f z;
#pragma unroll
  for (int i = 0; i < 8; ++i) z[i] = 0.0f;
  return z;
}

// ---------------- elementwise kernels ----------------

__global__ void k_scale(const float* __restrict__ in, float* __restrict__ out,
                        int n, float s) {
  int i = blockIdx.x * blockDim.x + threadIdx.x;
  if (i < n) out[i] = in[i] * s;
}

__global__ void k_cvt(const float* __restrict__ in, bf16_t* __restrict__ out, int n) {
  int i = blockIdx.x * blockDim.x + threadIdx.x;
  if (i < n) out[i] = (bf16_t)in[i];
}

__global__ void k_add(float* __restrict__ x, const float* __restrict__ d, int n) {
  int i = blockIdx.x * blockDim.x + threadIdx.x;
  if (i < n) x[i] += d[i];
}

__global__ void k_gelu_mul(const float* __restrict__ g, const float* __restrict__ u,
                           bf16_t* __restrict__ out, int n) {
  int i = blockIdx.x * blockDim.x + threadIdx.x;
  if (i < n) {
    float x = g[i];
    float t = tanhf(0.7978845608028654f * (x + 0.044715f * x * x * x));
    out[i] = (bf16_t)(0.5f * x * (1.0f + t) * u[i]);
  }
}

// ---------------- RMSNorm (block per row) ----------------

__global__ void k_rmsnorm(const float* __restrict__ x, const float* __restrict__ w,
                          bf16_t* __restrict__ out, int h) {
  int row = blockIdx.x;
  const float* xr = x + (long)row * h;
  float ss = 0.0f;
  for (int i = threadIdx.x; i < h; i += blockDim.x) { float v = xr[i]; ss += v * v; }
  __shared__ float red[256];
  red[threadIdx.x] = ss;
  __syncthreads();
  for (int s = 128; s > 0; s >>= 1) {
    if ((int)threadIdx.x < s) red[threadIdx.x] += red[threadIdx.x + s];
    __syncthreads();
  }
  float scale = rsqrtf(red[0] / (float)h + 1e-6f);
  for (int i = threadIdx.x; i < h; i += blockDim.x)
    out[(long)row * h + i] = (bf16_t)(xr[i] * scale * w[i]);
}

// ---------------- RoPE + bf16 repack of q/k/v ----------------
// q fp32 [B,T,NH*HD] -> qb bf16 [B,NH,T,HD] (rope applied)
// k fp32 [B,T,HD]    -> kb bf16 [B,T,HD]    (rope applied)
// v fp32 [B,T,HD]    -> vb bf16 [B,T,HD]
// position of row t is exactly t (streams are laid out at 0 / 512 / 513).

__global__ void k_rope(const float* __restrict__ qf, const float* __restrict__ kf,
                       const float* __restrict__ vf, bf16_t* __restrict__ qb,
                       bf16_t* __restrict__ kb, bf16_t* __restrict__ vb) {
  int t = blockIdx.x;          // 0..T-1
  int b = blockIdx.y;          // 0..B-1
  int i = threadIdx.x;         // 0..127  (HD/2)
  float ang = (float)t * powf(10000.0f, -((float)(2 * i)) / 256.0f);
  float c = cosf(ang), s = sinf(ang);

  long rk = ((long)b * T_ + t) * HD_;
  float k1 = kf[rk + i], k2 = kf[rk + i + 128];
  kb[rk + i]       = (bf16_t)(k1 * c - k2 * s);
  kb[rk + i + 128] = (bf16_t)(k2 * c + k1 * s);
  vb[rk + i]       = (bf16_t)vf[rk + i];
  vb[rk + i + 128] = (bf16_t)vf[rk + i + 128];

  long rq = ((long)b * T_ + t) * (NH_ * HD_);
#pragma unroll
  for (int hh = 0; hh < NH_; ++hh) {
    float q1 = qf[rq + hh * HD_ + i];
    float q2 = qf[rq + hh * HD_ + i + 128];
    long oo = (((long)(b * NH_ + hh)) * T_ + t) * HD_;
    qb[oo + i]       = (bf16_t)(q1 * c - q2 * s);
    qb[oo + i + 128] = (bf16_t)(q2 * c + q1 * s);
  }
}

// ---------------- softcap + block-mask + softmax ----------------

static __device__ inline int seg_of(int t) { return t < 512 ? 0 : (t < 513 ? 1 : 2); }

__global__ void k_softmax(const float* __restrict__ sc, bf16_t* __restrict__ pr) {
  int row = blockIdx.x;        // query index 0..T-1
  int z   = blockIdx.y;        // b*NH + h
  const float* srow = sc + ((long)z * T_ + row) * TP_;
  bf16_t*      prow = pr + ((long)z * T_ + row) * TP_;
  int sq = seg_of(row);

  __shared__ float buf[T_];
  __shared__ float red[256];

  float mx = -3.4e38f;
  for (int j = threadIdx.x; j < T_; j += blockDim.x) {
    float s = srow[j] * (1.0f / 16.0f);            // / sqrt(HD)
    s = tanhf(s * (1.0f / 50.0f)) * 50.0f;         // tanh softcap
    if (seg_of(j) > sq) s += -1e9f;                // block mask
    buf[j] = s;
    mx = fmaxf(mx, s);
  }
  red[threadIdx.x] = mx;
  __syncthreads();
  for (int s2 = 128; s2 > 0; s2 >>= 1) {
    if ((int)threadIdx.x < s2) red[threadIdx.x] = fmaxf(red[threadIdx.x], red[threadIdx.x + s2]);
    __syncthreads();
  }
  mx = red[0];
  __syncthreads();

  float sum = 0.0f;
  for (int j = threadIdx.x; j < T_; j += blockDim.x) {
    float e = expf(buf[j] - mx);
    buf[j] = e;
    sum += e;
  }
  red[threadIdx.x] = sum;
  __syncthreads();
  for (int s2 = 128; s2 > 0; s2 >>= 1) {
    if ((int)threadIdx.x < s2) red[threadIdx.x] += red[threadIdx.x + s2];
    __syncthreads();
  }
  float inv = 1.0f / red[0];
  for (int j = threadIdx.x; j < T_; j += blockDim.x)
    prow[j] = (bf16_t)(buf[j] * inv);
}

// ---------------- generic bf16 WMMA GEMM ----------------
// C[z] = A[z] (MxK, bf16) x B[z/zdiv] (KxN, TB) ; f32 accumulate/store.
// A(m,k) = A + m*lda + k ; B(k,n) = B + k*ldbK + n*ldbN ;
// C(m,n) = C + bIdx*cZ1 + (z%zdiv)*cZ2 + m*ldc + n.
// Block: 256 threads = 8 wave32. Block tile 128(M) x 64(N), BK=32.
// LDS double-buffered: As row-major [m][k], Bs transposed [n][k]; fragments
// are contiguous 16B/32B runs -> ds_load_b128.  Two barriers per K-step:
//   wait_asynccnt; barrier; async-stage tile i+1; WMMA on tile i; barrier.

template <typename TB>
__global__ __launch_bounds__(256) void k_gemm(
    const bf16_t* __restrict__ A, const TB* __restrict__ Bm, float* __restrict__ C,
    int M, int N, int K,
    long lda, long ldbK, long ldbN, long ldc,
    int zdiv, long aZ, long bZ, long cZ1, long cZ2) {
  int z = blockIdx.z;
  int bIdx = z / zdiv;
  A  += (long)z * aZ;
  Bm += (long)bIdx * bZ;
  C  += (long)bIdx * cZ1 + (long)(z - bIdx * zdiv) * cZ2;

  const int tid  = threadIdx.x;
  const int lane = tid & 31;
  const int wv   = tid >> 5;
  const int m0   = blockIdx.y * 128;
  const int n0   = blockIdx.x * 64;

  __shared__ alignas(16) bf16_t As[2][128 * 32];   // [m][k]
  __shared__ alignas(16) bf16_t Bs[2][64 * 32];    // [n][k]  (transposed)

  v8f acc[4];
#pragma unroll
  for (int i = 0; i < 4; ++i) acc[i] = zero_v8f();

  // B staging geometry: each thread fills 8 consecutive k at one column.
  const int bs_nn  = tid >> 2;          // 0..63 column
  const int bs_kk0 = (tid & 3) << 3;    // 0,8,16,24
  const int bs_gn  = n0 + bs_nn;

  auto stage = [&](int k0, bf16_t* Asb, bf16_t* Bsb) {
    // ---- A tile: 4096 bf16 = 512 x 16B chunks; 2 per thread ----
    if ((m0 + 128 <= M) && (k0 + 32 <= K)) {
#pragma unroll
      for (int e = 0; e < 2; ++e) {
        int chunk = tid * 2 + e;        // 0..511
        int r  = chunk >> 2;            // row in tile
        int kc = (chunk & 3) << 3;      // k offset (multiple of 8)
        const bf16_t* gp = A + (long)(m0 + r) * lda + (k0 + kc);
        bf16_t* lp = &Asb[r * 32 + kc];
#if USE_ASYNC_LDS
        async_copy16_to_lds(gp, lp);    // ASYNCcnt DMA, no VGPR round-trip
#else
        *(uint4*)lp = *(const uint4*)gp;
#endif
      }
    } else {
#pragma unroll
      for (int e = 0; e < 16; ++e) {
        int idx = tid * 16 + e;
        int r = idx >> 5, kk = idx & 31;
        int gm = m0 + r, gk = k0 + kk;
        Asb[idx] = (gm < M && gk < K) ? A[(long)gm * lda + gk] : (bf16_t)0.0f;
      }
    }
    // ---- B tile transposed [n][k]; one packed 16B LDS store per thread ----
    union { bf16_t h[8]; uint4 q; } pk;
    if (bs_gn < N && (k0 + 32 <= K)) {
      // unguarded straight-line loads (clause-friendly)
      const TB* q0 = Bm + (long)(k0 + bs_kk0) * ldbK + (long)bs_gn * ldbN;
#pragma unroll
      for (int e = 0; e < 8; ++e) pk.h[e] = (bf16_t)q0[(long)e * ldbK];
      // keep the streaming operand (weights) flowing into GL2 one tile ahead
      if (k0 + 64 <= K) __builtin_prefetch(q0 + 32 * ldbK, 0, 1);
    } else {
#pragma unroll
      for (int e = 0; e < 8; ++e) {
        int gk = k0 + bs_kk0 + e;
        pk.h[e] = (gk < K && bs_gn < N)
                      ? (bf16_t)Bm[(long)gk * ldbK + (long)bs_gn * ldbN]
                      : (bf16_t)0.0f;
      }
    }
    *(uint4*)&Bsb[bs_nn * 32 + bs_kk0] = pk.q;
  };

  const int nk = (K + 31) >> 5;
  stage(0, As[0], Bs[0]);

  for (int i = 0; i < nk; ++i) {
    wait_asynccnt0();            // our async stage of tile i has landed
    __syncthreads();             // everyone's tile i visible; buffers released
    if (i + 1 < nk) stage(32 * (i + 1), As[(i + 1) & 1], Bs[(i + 1) & 1]);

    const bf16_t* Asb = As[i & 1];
    const bf16_t* Bsb = Bs[i & 1];

    // A fragment (ISA 7.12.2, 16-bit A 16x32): lane&15 = row, lane bit4 picks
    // K-octet {0..7,16..23} vs {8..15,24..31}.  Two 16B LDS loads.
    union { v16bf f; uint4 q[2]; } au;
    {
      int r  = (wv << 4) + (lane & 15);
      int kb = (lane & 16) ? 8 : 0;
      const bf16_t* ap = &Asb[r * 32 + kb];
      au.q[0] = *(const uint4*)(ap);
      au.q[1] = *(const uint4*)(ap + 16);
    }
    // B fragments: lane&15 = column, lane bit4 picks K-octet {0..7,16..23} vs
    // {8..15,24..31} (same as A); two 16B runs from the transposed LDS layout.
    int kb2 = (lane & 16) ? 8 : 0;
    int cc  = lane & 15;
#pragma unroll
    for (int nt = 0; nt < 4; ++nt) {
      union { v16bf f; uint4 q[2]; } bu;
      const bf16_t* bp = &Bsb[(nt * 16 + cc) * 32 + kb2];
      bu.q[0] = *(const uint4*)(bp);
      bu.q[1] = *(const uint4*)(bp + 16);
      acc[nt] = __builtin_amdgcn_wmma_f32_16x16x32_bf16(
          false, au.f, false, bu.f, (short)0, acc[nt], false, false);
    }
    __syncthreads();             // all reads of tile i done before it is restaged
  }

  // C fragment (ISA 7.12.2, 32-bit C 16x16): vreg f -> M = f + 8*lane[4], N = lane&15
  int rbase = m0 + (wv << 4) + ((lane & 16) ? 8 : 0);
  int cbase = n0 + (lane & 15);
#pragma unroll
  for (int nt = 0; nt < 4; ++nt) {
#pragma unroll
    for (int f = 0; f < 8; ++f) {
      int gm = rbase + f;
      int gn = cbase + nt * 16;
      if (gm < M && gn < N) C[(long)gm * ldc + gn] = acc[nt][f];
    }
  }
}

// ---------------------------------------------------------------------------
// host driver
// ---------------------------------------------------------------------------

static inline int cdiv_i(int a, int b) { return (a + b - 1) / b; }

extern "C" void kernel_launch(void* const* d_in, const int* in_sizes, int n_in,
                              void* d_out, int out_size, void* d_ws, size_t ws_size,
                              hipStream_t stream) {
  (void)in_sizes; (void)n_in; (void)out_size; (void)ws_size;

  // setup_inputs() order:
  // 0 attention_mask (unused; mask is analytic), 1-3 pos_* (unused; pos == row),
  // 4-6 emb_{vlm,proprio,action},
  // 7..15 p_vlm {Wq,Wk,Wv,Wo,Wg,Wu,Wd,ln1,ln2}, 16..24 p_proprio, 25..33 p_action
  const float* emb[3] = {(const float*)d_in[4], (const float*)d_in[5],
                         (const float*)d_in[6]};
  struct P { const float *Wq, *Wk, *Wv, *Wo, *Wg, *Wu, *Wd, *ln1, *ln2; } p[3];
  for (int n = 0; n < 3; ++n) {
    int b = 7 + 9 * n;
    p[n].Wq  = (const float*)d_in[b + 0];
    p[n].Wk  = (const float*)d_in[b + 1];
    p[n].Wv  = (const float*)d_in[b + 2];
    p[n].Wo  = (const float*)d_in[b + 3];
    p[n].Wg  = (const float*)d_in[b + 4];
    p[n].Wu  = (const float*)d_in[b + 5];
    p[n].Wd  = (const float*)d_in[b + 6];
    p[n].ln1 = (const float*)d_in[b + 7];
    p[n].ln2 = (const float*)d_in[b + 8];
  }

  const int H[3]  = {2048, 1024, 1024};
  const int IT[3] = {8192, 4096, 4096};
  const int S[3]  = {512, 1, 32};
  const int OFF[3] = {0, 512, 513};

  // outputs live directly in d_out (tuple: vlm | proprio | action, flat)
  float* xs[3];
  xs[0] = (float*)d_out;                       // 2*512*2048
  xs[1] = (float*)d_out + 2097152;             // 2*1*1024
  xs[2] = (float*)d_out + 2099200;             // 2*32*1024

  // workspace bump allocator
  char* wsp = (char*)d_ws;
  auto alloc = [&](size_t bytes) -> void* {
    void* r = (void*)wsp;
    wsp += (bytes + 255) & ~(size_t)255;
    return r;
  };
  bf16_t* hbf = (bf16_t*)alloc((size_t)1024 * 2048 * 2);            // normalized (max)
  float*  qf  = (float*) alloc((size_t)B_ * T_ * 2048 * 4);
  float*  kf  = (float*) alloc((size_t)B_ * T_ * HD_ * 4);
  float*  vf  = (float*) alloc((size_t)B_ * T_ * HD_ * 4);
  bf16_t* qb  = (bf16_t*)alloc((size_t)B_ * NH_ * T_ * HD_ * 2);
  bf16_t* kb  = (bf16_t*)alloc((size_t)B_ * T_ * HD_ * 2);
  bf16_t* vb  = (bf16_t*)alloc((size_t)B_ * T_ * HD_ * 2);
  float*  sc  = (float*) alloc((size_t)B_ * NH_ * T_ * TP_ * 4);
  bf16_t* pr  = (bf16_t*)alloc((size_t)B_ * NH_ * T_ * TP_ * 2);
  float*  oatt= (float*) alloc((size_t)B_ * T_ * 2048 * 4);
  bf16_t* ob  = (bf16_t*)alloc((size_t)B_ * T_ * 2048 * 2);
  float*  tmp = (float*) alloc((size_t)1024 * 2048 * 4);            // proj out (max)
  float*  g   = (float*) alloc((size_t)2 * 512 * 8192 * 4);
  float*  u   = (float*) alloc((size_t)2 * 512 * 8192 * 4);
  bf16_t* act = (bf16_t*)alloc((size_t)2 * 512 * 8192 * 2);

  auto gemm_f = [&](const bf16_t* A, const float* Bw, float* Cc, int M, int N, int K,
                    long lda, long ldbK, long ldbN, long ldc, int zdiv,
                    long aZ, long bZ, long cZ1, long cZ2, int Z) {
    dim3 grid(cdiv_i(N, 64), cdiv_i(M, 128), Z);
    k_gemm<float><<<grid, 256, 0, stream>>>(A, Bw, Cc, M, N, K, lda, ldbK, ldbN,
                                            ldc, zdiv, aZ, bZ, cZ1, cZ2);
  };
  auto gemm_b = [&](const bf16_t* A, const bf16_t* Bw, float* Cc, int M, int N, int K,
                    long lda, long ldbK, long ldbN, long ldc, int zdiv,
                    long aZ, long bZ, long cZ1, long cZ2, int Z) {
    dim3 grid(cdiv_i(N, 64), cdiv_i(M, 128), Z);
    k_gemm<bf16_t><<<grid, 256, 0, stream>>>(A, Bw, Cc, M, N, K, lda, ldbK, ldbN,
                                             ldc, zdiv, aZ, bZ, cZ1, cZ2);
  };

  // 1) x = emb * sqrt(h)
  for (int n = 0; n < 3; ++n) {
    int ne = B_ * S[n] * H[n];
    k_scale<<<cdiv_i(ne, 256), 256, 0, stream>>>(emb[n], xs[n], ne,
                                                 sqrtf((float)H[n]));
  }

  for (int l = 0; l < L_; ++l) {
    // ---- attention: per-stream QKV projections ----
    for (int n = 0; n < 3; ++n) {
      int rows = B_ * S[n];
      k_rmsnorm<<<rows, 256, 0, stream>>>(xs[n], p[n].ln1 + (size_t)l * H[n], hbf, H[n]);
      long aZ = (long)S[n] * H[n];
      // Q: [S,h] x [h,2048] -> qf rows at (b*T + OFF + s)
      gemm_f(hbf, p[n].Wq + (size_t)l * H[n] * 2048, qf + (size_t)OFF[n] * 2048,
             S[n], 2048, H[n], H[n], 2048, 1, 2048, 1, aZ, 0, (long)T_ * 2048, 0, B_);
      // K
      gemm_f(hbf, p[n].Wk + (size_t)l * H[n] * HD_, kf + (size_t)OFF[n] * HD_,
             S[n], HD_, H[n], H[n], HD_, 1, HD_, 1, aZ, 0, (long)T_ * HD_, 0, B_);
      // V
      gemm_f(hbf, p[n].Wv + (size_t)l * H[n] * HD_, vf + (size_t)OFF[n] * HD_,
             S[n], HD_, H[n], H[n], HD_, 1, HD_, 1, aZ, 0, (long)T_ * HD_, 0, B_);
    }

    // ---- RoPE + bf16 repack ----
    k_rope<<<dim3(T_, B_), 128, 0, stream>>>(qf, kf, vf, qb, kb, vb);

    // ---- scores = Q x K^T : z = b*NH+h, KV shared per batch (zdiv=8) ----
    gemm_b(qb, kb, sc, T_, T_, HD_,
           HD_, 1, HD_, TP_, NH_,
           (long)T_ * HD_, (long)T_ * HD_, (long)NH_ * T_ * TP_, (long)T_ * TP_,
           B_ * NH_);

    // ---- softcap + mask + softmax -> bf16 probs ----
    k_softmax<<<dim3(T_, B_ * NH_), 256, 0, stream>>>(sc, pr);

    // ---- O = P x V -> oatt [B,T,NH*HD] (head h at cols h*256..) ----
    gemm_b(pr, vb, oatt, T_, HD_, T_,
           TP_, HD_, 1, 2048, NH_,
           (long)T_ * TP_, (long)T_ * HD_, (long)T_ * 2048, (long)HD_,
           B_ * NH_);

    int no = B_ * T_ * 2048;
    k_cvt<<<cdiv_i(no, 256), 256, 0, stream>>>(oatt, ob, no);

    // ---- per-stream: Wo proj + residual, then GeGLU MLP + residual ----
    for (int n = 0; n < 3; ++n) {
      int rows = B_ * S[n];
      int nres = rows * H[n];
      // attn = o[:, off:off+S] @ Wo
      gemm_f(ob + (size_t)OFF[n] * 2048, p[n].Wo + (size_t)l * 2048 * H[n], tmp,
             S[n], H[n], 2048, 2048, H[n], 1, H[n], 1,
             (long)T_ * 2048, 0, (long)S[n] * H[n], 0, B_);
      k_add<<<cdiv_i(nres, 256), 256, 0, stream>>>(xs[n], tmp, nres);

      // MLP
      k_rmsnorm<<<rows, 256, 0, stream>>>(xs[n], p[n].ln2 + (size_t)l * H[n], hbf, H[n]);
      long aZ = (long)S[n] * H[n];
      gemm_f(hbf, p[n].Wg + (size_t)l * H[n] * IT[n], g,
             S[n], IT[n], H[n], H[n], IT[n], 1, IT[n], 1,
             aZ, 0, (long)S[n] * IT[n], 0, B_);
      gemm_f(hbf, p[n].Wu + (size_t)l * H[n] * IT[n], u,
             S[n], IT[n], H[n], H[n], IT[n], 1, IT[n], 1,
             aZ, 0, (long)S[n] * IT[n], 0, B_);
      int nit = rows * IT[n];
      k_gelu_mul<<<cdiv_i(nit, 256), 256, 0, stream>>>(g, u, act, nit);
      gemm_f(act, p[n].Wd + (size_t)l * IT[n] * H[n], tmp,
             S[n], H[n], IT[n], IT[n], H[n], 1, H[n], 1,
             (long)S[n] * IT[n], 0, (long)S[n] * H[n], 0, B_);
      k_add<<<cdiv_i(nres, 256), 256, 0, stream>>>(xs[n], tmp, nres);
    }
  }
}
